// SelfAttention_71863392797501
// MI455X (gfx1250) — matrix operand order repaired and checked by reading, not compiled, not run on hardware
//
#include <hip/hip_runtime.h>
#include <hip/hip_bf16.h>

// ---------------------------------------------------------------------------
// Self-attention forward for MI455X (gfx1250, wave32, WMMA + TDM).
// B=2, N=2048, DIM=1024, HEADS=16, DIM_HEAD=64.
//   1) ln_kernel:   h = LayerNorm(x) -> f16                [4096 x 1024]
//   2) cvt_kernel:  w_qkv, w_out -> f16
//   3) qkv_gemm:    h @ w_qkv^T via TDM-staged LDS tiles, scatter Q/K/V^T
//   4) attn_kernel: flash attention, TDM-staged K/V tiles, v_wmma f16
//   5) out_gemm:    Ow @ w_out^T via TDM-staged LDS tiles -> f32 d_out
// ---------------------------------------------------------------------------

typedef _Float16 v16h __attribute__((ext_vector_type(16)));
typedef _Float16 v8h  __attribute__((ext_vector_type(8)));
typedef float    v8f  __attribute__((ext_vector_type(8)));
typedef unsigned int u32x4 __attribute__((ext_vector_type(4)));
typedef int          i32x4v __attribute__((ext_vector_type(4)));
typedef int          i32x8v __attribute__((ext_vector_type(8)));

#define BDIM   2
#define SEQ    2048
#define DIM    1024
#define HEADS  16
#define DHEAD  64
#define NEGMAX (-3.402823466e38f)

#if __has_include(<hip/amd_detail/amd_gfx1250_TDM.h>)
#define TDM_SIX_ARGS 1
#endif

// ---- Tensor Data Mover helpers -------------------------------------------
__device__ inline unsigned lds_off_u32(const void* p) {
  return (unsigned)(unsigned long long)(__attribute__((address_space(3))) const void*)p;
}

__device__ inline void tdm_raw(u32x4 g0, i32x8v g1) {
#if defined(__HIP_DEVICE_COMPILE__)
#if __has_builtin(__builtin_amdgcn_tensor_load_to_lds)
  i32x4v z4 = {0, 0, 0, 0};
#if defined(TDM_SIX_ARGS)
  i32x8v z8 = {0, 0, 0, 0, 0, 0, 0, 0};
  __builtin_amdgcn_tensor_load_to_lds(g0, g1, z4, z4, z8, 0);
#else
  __builtin_amdgcn_tensor_load_to_lds(g0, g1, z4, z4, 0);
#endif
#endif
#endif
}

// 2-D f16 tile load. Tensor rows have dim0 elements (== row stride), dim1 rows.
// Tile: tile0 contiguous elements x tile1 rows, packed contiguously into LDS.
__device__ inline void tdm_tile_2d(const _Float16* gtile, const _Float16* lds_dst,
                                   unsigned dim0, unsigned dim1,
                                   unsigned tile0, unsigned tile1) {
  unsigned long long ga = (unsigned long long)(size_t)gtile;
  u32x4 g0;
  g0.x = 1u;                                                  // count=1, user-mode D#
  g0.y = lds_off_u32(lds_dst);                                // lds_addr (bytes)
  g0.z = (unsigned)(ga & 0xffffffffull);                      // global_addr[31:0]
  g0.w = (unsigned)((ga >> 32) & 0x01ffffffull) | (2u << 30); // addr[56:32] | type=2
  i32x8v g1;
  g1[0] = 0x00010000;                                         // data_size=1 (2 bytes)
  g1[1] = (int)((dim0 & 0xffffu) << 16);                      // tensor_dim0[15:0]
  g1[2] = (int)(((dim0 >> 16) & 0xffffu) | ((dim1 & 0xffffu) << 16));
  g1[3] = (int)(((dim1 >> 16) & 0xffffu) | (tile0 << 16));    // dim1 hi | tile_dim0
  g1[4] = (int)tile1;                                         // tile_dim1, tile_dim2=0
  g1[5] = (int)dim0;                                          // tensor_dim0_stride lo32
  g1[6] = 0;
  g1[7] = 0;
  tdm_raw(g0, g1);
}

__device__ inline void tensor_wait0() {
#if defined(__HIP_DEVICE_COMPILE__)
#if __has_builtin(__builtin_amdgcn_s_wait_tensorcnt)
  __builtin_amdgcn_s_wait_tensorcnt(0);
#else
  asm volatile("s_wait_tensorcnt 0" ::: "memory");
#endif
#endif
}

// ---- WMMA fragment loaders (ISA 16-bit A 16x32 / B 32x16 layouts) --------
__device__ inline v16h frag_a16(const _Float16* __restrict__ p) {
  v8h lo = *(const v8h*)(p);        // k[0..7] (or 8..15 for upper half-wave)
  v8h hi = *(const v8h*)(p + 16);   // k[16..23] (or 24..31)
  v16h r;
#pragma unroll
  for (int i = 0; i < 8; ++i) { r[i] = lo[i]; r[8 + i] = hi[i]; }
  return r;
}

__device__ inline v16h frag_b16(const _Float16* __restrict__ p) {
  v8h lo = *(const v8h*)(p);        // k[0..7] (or 8..15 for upper half-wave)
  v8h hi = *(const v8h*)(p + 16);   // k[16..23] (or 24..31)
  v16h r;
#pragma unroll
  for (int i = 0; i < 8; ++i) { r[i] = lo[i]; r[8 + i] = hi[i]; }
  return r;
}

// --------------------------- 1) LayerNorm -> f16 ---------------------------
__global__ void __launch_bounds__(256)
ln_kernel(const float* __restrict__ x, const float* __restrict__ w,
          const float* __restrict__ bias, _Float16* __restrict__ h) {
  const int row = blockIdx.x;
  const float* xr = x + (size_t)row * DIM;
  float s = 0.f, ss = 0.f;
  for (int i = threadIdx.x; i < DIM; i += 256) { float v = xr[i]; s += v; ss += v * v; }
#pragma unroll
  for (int off = 1; off < 32; off <<= 1) {
    s  += __shfl_xor(s,  off, 32);
    ss += __shfl_xor(ss, off, 32);
  }
  __shared__ float sw[8], ssw[8];
  if ((threadIdx.x & 31) == 0) { sw[threadIdx.x >> 5] = s; ssw[threadIdx.x >> 5] = ss; }
  __syncthreads();
  float S = 0.f, SS = 0.f;
#pragma unroll
  for (int i = 0; i < 8; ++i) { S += sw[i]; SS += ssw[i]; }
  const float mu  = S * (1.f / DIM);
  const float var = SS * (1.f / DIM) - mu * mu;
  const float inv = rsqrtf(var + 1e-5f);
  for (int i = threadIdx.x; i < DIM; i += 256)
    h[(size_t)row * DIM + i] = (_Float16)((xr[i] - mu) * inv * w[i] + bias[i]);
}

// --------------------------- 2) f32 -> f16 convert -------------------------
__global__ void __launch_bounds__(256)
cvt_kernel(const float* __restrict__ src, _Float16* __restrict__ dst, int n) {
  int i = blockIdx.x * 256 + threadIdx.x;
  if (i < n) dst[i] = (_Float16)src[i];
}

// --------------------------- 3) QKV projection -----------------------------
// C[4096 x 3072] = h[4096 x 1024] @ w^T. Block 256 thr = 8 waves,
// tile 128M x 128N, TDM double-buffered LDS staging, wave = 32M x 64N.
__global__ void __launch_bounds__(256)
qkv_gemm(const _Float16* __restrict__ h, const _Float16* __restrict__ w,
         _Float16* __restrict__ Qb, _Float16* __restrict__ Kb,
         _Float16* __restrict__ Vt) {
  __shared__ _Float16 Ab[2][128 * 32];
  __shared__ _Float16 Bb[2][128 * 32];
  const int lane = threadIdx.x & 31;
  const int wid  = threadIdx.x >> 5;
  const int wm   = wid & 3;            // 4 M waves (32 rows each)
  const int wn   = wid >> 2;           // 2 N waves (64 cols each)
  const int mBlk = blockIdx.y * 128;
  const int nBlk = blockIdx.x * 128;

  const int rA    = lane & 15;
  const int kb8   = (lane >> 4) << 3;
  const int kb16  = (lane >> 4) << 4;

  v8f c[2][4];
#pragma unroll
  for (int s = 0; s < 2; ++s)
#pragma unroll
    for (int t = 0; t < 4; ++t) c[s][t] = v8f{};

  auto stage = [&](int buf, int kk) {
    if (wid == 0) {
      tdm_tile_2d(h + (size_t)mBlk * DIM + kk, &Ab[buf][0], DIM, 4096u, 32u, 128u);
      tdm_tile_2d(w + (size_t)nBlk * DIM + kk, &Bb[buf][0], DIM, 3072u, 32u, 128u);
    }
  };

  int buf = 0;
  stage(buf, 0);
  if (wid == 0) tensor_wait0();
  __syncthreads();

  for (int kk = 0; kk < DIM; kk += 32) {
    if (kk + 32 < DIM) stage(buf ^ 1, kk + 32);

    const _Float16* As = &Ab[buf][0];
    const _Float16* Bs = &Bb[buf][0];
    const v16h a0 = frag_a16(As + (wm * 32 + rA) * 32 + kb8);
    const v16h a1 = frag_a16(As + (wm * 32 + 16 + rA) * 32 + kb8);
#pragma unroll
    for (int t = 0; t < 4; ++t) {
      v16h bf = frag_b16(Bs + (wn * 64 + t * 16 + rA) * 32 + kb8);
      c[0][t] = __builtin_amdgcn_wmma_f32_16x16x32_f16(false, a0, false, bf,
                                                       (short)0, c[0][t], false, false);
      c[1][t] = __builtin_amdgcn_wmma_f32_16x16x32_f16(false, a1, false, bf,
                                                       (short)0, c[1][t], false, false);
    }

    if (kk + 32 < DIM) {
      if (wid == 0) tensor_wait0();
      __syncthreads();
      buf ^= 1;
    }
  }

  // ---- scatter to Q (*scale) / K / V^T ----------------------------------
  const int colL   = lane & 15;
  const int rowOff = (lane >> 4) * 8;
#pragma unroll
  for (int s = 0; s < 2; ++s) {
#pragma unroll
    for (int t = 0; t < 4; ++t) {
      const int gcol  = nBlk + wn * 64 + t * 16 + colL;
      const int which = gcol >> 10;          // 0=q 1=k 2=v (uniform per tile)
      const int hc    = (gcol & 1023) >> 6;
      const int dc    = gcol & 63;
#pragma unroll
      for (int j = 0; j < 8; ++j) {
        const int grow = mBlk + wm * 32 + s * 16 + rowOff + j;
        const int bb = grow >> 11;
        const int nn = grow & 2047;
        const float v = c[s][t][j];
        if (which == 0)
          Qb[(((size_t)(bb * HEADS + hc) * SEQ) + nn) * DHEAD + dc] = (_Float16)(v * 0.125f);
        else if (which == 1)
          Kb[(((size_t)(bb * HEADS + hc) * SEQ) + nn) * DHEAD + dc] = (_Float16)v;
        else
          Vt[((size_t)(bb * HEADS + hc) * DHEAD + dc) * SEQ + nn] = (_Float16)v;
      }
    }
  }
}

// --------------------------- 4) Flash attention ----------------------------
// Grid: (SEQ/128, B*H). Block: 256 thr = 8 waves; wave owns a 16-row q tile.
// K/V tiles for each 32-key block are TDM-staged into LDS, shared by 8 waves.
__global__ void __launch_bounds__(256)
attn_kernel(const _Float16* __restrict__ Qb, const _Float16* __restrict__ Kb,
            const _Float16* __restrict__ Vt, const unsigned char* __restrict__ mask,
            _Float16* __restrict__ Ow) {
  const int lane = threadIdx.x & 31;
  const int wid  = threadIdx.x >> 5;
  const int bh   = blockIdx.y;
  const int b    = bh >> 4;
  const int hh   = bh & 15;
  const int qBase = blockIdx.x * 128 + wid * 16;

  __shared__ _Float16 Kt[2][32 * DHEAD];   // [key][d]   4 KB per buffer
  __shared__ _Float16 Vs[2][DHEAD * 32];   // [d][key]   4 KB per buffer
  __shared__ _Float16 plds[8][16 * 32];    // per-wave P transpose staging
  _Float16* pw = &plds[wid][0];

  const int rA   = lane & 15;
  const int kb8  = (lane >> 4) << 3;
  const int kb16 = (lane >> 4) << 4;
  const int rowOff = (lane >> 4) * 8;

  const _Float16* Qrow = Qb + ((size_t)bh * SEQ + qBase + rA) * DHEAD + kb8;
  const v16h aq0 = frag_a16(Qrow);
  const v16h aq1 = frag_a16(Qrow + 32);

  const _Float16* Kbh = Kb + (size_t)bh * SEQ * DHEAD;
  const _Float16* Vbh = Vt + (size_t)bh * DHEAD * SEQ;
  const unsigned char* mb = mask + (size_t)b * SEQ * SEQ;

  v8f o[4] = {v8f{}, v8f{}, v8f{}, v8f{}};
  float mrow[8], lrow[8];
#pragma unroll
  for (int j = 0; j < 8; ++j) { mrow[j] = NEGMAX; lrow[j] = 0.f; }

  auto stage = [&](int buf, int k0) {
    if (wid == 0) {
      // K tensor [SEQ x 64]: tile = 32 key-rows x 64 d (contiguous rows)
      tdm_tile_2d(Kbh + (size_t)k0 * DHEAD, &Kt[buf][0], (unsigned)DHEAD, (unsigned)SEQ,
                  (unsigned)DHEAD, 32u);
      // V^T tensor [64 x SEQ]: tile = 32 contiguous keys x 64 d-rows
      tdm_tile_2d(Vbh + k0, &Vs[buf][0], (unsigned)SEQ, (unsigned)DHEAD, 32u,
                  (unsigned)DHEAD);
    }
  };

  int buf = 0;
  stage(buf, 0);
  if (wid == 0) tensor_wait0();
  __syncthreads();

  for (int k0 = 0; k0 < SEQ; k0 += 32) {
    if (k0 + 32 < SEQ) stage(buf ^ 1, k0 + 32);

    // ---- S = Q K^T for 32 keys (two 16x16 C tiles) from LDS K tile ----
    const _Float16* Ks = &Kt[buf][0];
    v8f s0 = {}, s1 = {};
    {
      const _Float16* kc0 = Ks + (size_t)rA * DHEAD + kb8;         // keys 0..15
      v16h bk = frag_b16(kc0);
      s0 = __builtin_amdgcn_wmma_f32_16x16x32_f16(false, aq0, false, bk, (short)0, s0, false, false);
      bk = frag_b16(kc0 + 32);
      s0 = __builtin_amdgcn_wmma_f32_16x16x32_f16(false, aq1, false, bk, (short)0, s0, false, false);
      const _Float16* kc1 = kc0 + 16 * DHEAD;                      // keys 16..31
      bk = frag_b16(kc1);
      s1 = __builtin_amdgcn_wmma_f32_16x16x32_f16(false, aq0, false, bk, (short)0, s1, false, false);
      bk = frag_b16(kc1 + 32);
      s1 = __builtin_amdgcn_wmma_f32_16x16x32_f16(false, aq1, false, bk, (short)0, s1, false, false);
    }

    // ---- mask + online softmax (rows striped over 16-lane groups) ----
    const int key0 = k0 + rA;
    __builtin_prefetch(mb + (size_t)(qBase + rowOff) * SEQ + key0 + 32, 0, 1);
    float p0[8], p1[8];
#pragma unroll
    for (int j = 0; j < 8; ++j) {
      const int qr = qBase + rowOff + j;
      const unsigned char* mp = mb + (size_t)qr * SEQ + key0;
      float sv0 = mp[0]  ? NEGMAX : s0[j];
      float sv1 = mp[16] ? NEGMAX : s1[j];
      float mx = fmaxf(sv0, sv1);
#pragma unroll
      for (int off = 1; off < 16; off <<= 1) mx = fmaxf(mx, __shfl_xor(mx, off, 32));
      const float mn    = fmaxf(mrow[j], mx);
      const float alpha = __expf(mrow[j] - mn);
      mrow[j] = mn;
      const float e0 = __expf(sv0 - mn);
      const float e1 = __expf(sv1 - mn);
      float sum = e0 + e1;
#pragma unroll
      for (int off = 1; off < 16; off <<= 1) sum += __shfl_xor(sum, off, 32);
      lrow[j] = lrow[j] * alpha + sum;
      o[0][j] *= alpha; o[1][j] *= alpha; o[2][j] *= alpha; o[3][j] *= alpha;
      p0[j] = e0; p1[j] = e1;
    }

    // ---- transpose P (C layout -> A fragment) through per-wave LDS ----
#pragma unroll
    for (int j = 0; j < 8; ++j) {
      pw[(rowOff + j) * 32 + rA]      = (_Float16)p0[j];
      pw[(rowOff + j) * 32 + 16 + rA] = (_Float16)p1[j];
    }
    asm volatile("s_wait_dscnt 0" ::: "memory");
    const v16h pa = frag_a16(pw + rA * 32 + kb8);

    // ---- O += P @ V from LDS V tile ----
    const _Float16* Vsb = &Vs[buf][0];
#pragma unroll
    for (int td = 0; td < 4; ++td) {
      v16h bv = frag_b16(Vsb + (size_t)(td * 16 + rA) * 32 + kb8);
      o[td] = __builtin_amdgcn_wmma_f32_16x16x32_f16(false, pa, false, bv,
                                                     (short)0, o[td], false, false);
    }

    if (k0 + 32 < SEQ) {
      if (wid == 0) tensor_wait0();
      __syncthreads();
      buf ^= 1;
    }
  }

  // ---- normalize and store Ow[b][n][h*64 + d] ----
#pragma unroll
  for (int j = 0; j < 8; ++j) {
    const float inv = 1.f / lrow[j];
    const int qr = qBase + rowOff + j;
    _Float16* orow = Ow + ((size_t)b * SEQ + qr) * DIM + hh * DHEAD + rA;
#pragma unroll
    for (int td = 0; td < 4; ++td)
      orow[td * 16] = (_Float16)(o[td][j] * inv);
  }
}

// --------------------------- 5) Output projection --------------------------
// out[4096 x 1024] f32 = Ow[4096 x 1024] f16 @ w_out^T, TDM-staged LDS tiles.
__global__ void __launch_bounds__(256)
out_gemm(const _Float16* __restrict__ Ow, const _Float16* __restrict__ w,
         float* __restrict__ out) {
  __shared__ _Float16 Ab[2][128 * 32];
  __shared__ _Float16 Bb[2][128 * 32];
  const int lane = threadIdx.x & 31;
  const int wid  = threadIdx.x >> 5;
  const int wm   = wid & 3;
  const int wn   = wid >> 2;
  const int mBlk = blockIdx.y * 128;
  const int nBlk = blockIdx.x * 128;

  const int rA   = lane & 15;
  const int kb8  = (lane >> 4) << 3;
  const int kb16 = (lane >> 4) << 4;

  v8f c[2][4];
#pragma unroll
  for (int s = 0; s < 2; ++s)
#pragma unroll
    for (int t = 0; t < 4; ++t) c[s][t] = v8f{};

  auto stage = [&](int buf, int kk) {
    if (wid == 0) {
      tdm_tile_2d(Ow + (size_t)mBlk * DIM + kk, &Ab[buf][0], DIM, 4096u, 32u, 128u);
      tdm_tile_2d(w  + (size_t)nBlk * DIM + kk, &Bb[buf][0], DIM, 1024u, 32u, 128u);
    }
  };

  int buf = 0;
  stage(buf, 0);
  if (wid == 0) tensor_wait0();
  __syncthreads();

  for (int kk = 0; kk < DIM; kk += 32) {
    if (kk + 32 < DIM) stage(buf ^ 1, kk + 32);

    const _Float16* As = &Ab[buf][0];
    const _Float16* Bs = &Bb[buf][0];
    const v16h a0 = frag_a16(As + (wm * 32 + rA) * 32 + kb8);
    const v16h a1 = frag_a16(As + (wm * 32 + 16 + rA) * 32 + kb8);
#pragma unroll
    for (int t = 0; t < 4; ++t) {
      v16h bf = frag_b16(Bs + (wn * 64 + t * 16 + rA) * 32 + kb8);
      c[0][t] = __builtin_amdgcn_wmma_f32_16x16x32_f16(false, a0, false, bf,
                                                       (short)0, c[0][t], false, false);
      c[1][t] = __builtin_amdgcn_wmma_f32_16x16x32_f16(false, a1, false, bf,
                                                       (short)0, c[1][t], false, false);
    }

    if (kk + 32 < DIM) {
      if (wid == 0) tensor_wait0();
      __syncthreads();
      buf ^= 1;
    }
  }

  const int colL   = lane & 15;
  const int rowOff = (lane >> 4) * 8;
#pragma unroll
  for (int s = 0; s < 2; ++s) {
#pragma unroll
    for (int t = 0; t < 4; ++t) {
      const int gcol = nBlk + wn * 64 + t * 16 + colL;
#pragma unroll
      for (int j = 0; j < 8; ++j) {
        const int grow = mBlk + wm * 32 + s * 16 + rowOff + j;
        out[(size_t)grow * DIM + gcol] = c[s][t][j];
      }
    }
  }
}

// ---------------------------------------------------------------------------
extern "C" void kernel_launch(void* const* d_in, const int* in_sizes, int n_in,
                              void* d_out, int out_size, void* d_ws, size_t ws_size,
                              hipStream_t stream) {
  const float* x     = (const float*)d_in[0];
  const unsigned char* mask = (const unsigned char*)d_in[1];   // jnp.bool_ = 1 byte
  const float* ln_w  = (const float*)d_in[2];
  const float* ln_b  = (const float*)d_in[3];
  const float* w_qkv = (const float*)d_in[4];
  const float* w_out = (const float*)d_in[5];
  float* out = (float*)d_out;

  const size_t ROWS = (size_t)BDIM * SEQ;          // 4096
  _Float16* ws  = (_Float16*)d_ws;
  _Float16* h   = ws;                              // 4096*1024
  _Float16* wq  = h  + ROWS * DIM;                 // 3072*1024
  _Float16* wo  = wq + (size_t)3 * DIM * DIM;      // 1024*1024
  _Float16* Qb  = wo + (size_t)DIM * DIM;          // [B,H,N,d]
  _Float16* Kb  = Qb + (size_t)BDIM * HEADS * SEQ * DHEAD;
  _Float16* Vt  = Kb + (size_t)BDIM * HEADS * SEQ * DHEAD;   // [B,H,d,N]
  _Float16* Ob  = Vt + (size_t)BDIM * HEADS * SEQ * DHEAD;   // [B,N,INNER]

  ln_kernel<<<(int)ROWS, 256, 0, stream>>>(x, ln_w, ln_b, h);

  const int nqkv = 3 * DIM * DIM;
  cvt_kernel<<<(nqkv + 255) / 256, 256, 0, stream>>>(w_qkv, wq, nqkv);
  const int nout = DIM * DIM;
  cvt_kernel<<<(nout + 255) / 256, 256, 0, stream>>>(w_out, wo, nout);

  qkv_gemm<<<dim3(3 * DIM / 128, (int)ROWS / 128), 256, 0, stream>>>(h, wq, Qb, Kb, Vt);

  attn_kernel<<<dim3(SEQ / 128, BDIM * HEADS), 256, 0, stream>>>(Qb, Kb, Vt, mask, Ob);

  out_gemm<<<dim3(DIM / 128, (int)ROWS / 128), 256, 0, stream>>>(Ob, wo, out);
}
